// LSTMModel_35124242546852
// MI455X (gfx1250) — matrix operand order repaired and checked by reading, not compiled, not run on hardware
//
#include <hip/hip_runtime.h>
#include <hip/hip_bf16.h>

// ---------------------------------------------------------------------------
// 2-layer LSTM (H=50, B=4096, T=512, D=1) + final FC, fused per 16-batch tile.
// Recurrent GEMMs mapped to v_wmma_f32_16x16x32_bf16 with H padded 50->64 and
// gate dim padded 200->256 (padding rows zeroed -> numerically inert).
// Compute-bound problem (~126 GFLOP, ~8MB traffic): WMMA + LDS exchange only.
// ---------------------------------------------------------------------------

typedef __attribute__((ext_vector_type(16))) __bf16 v16bf;
typedef __attribute__((ext_vector_type(8)))  float  v8f;
typedef __attribute__((ext_vector_type(4)))  float  v4f;

union BF16x16 { v16bf v; v4f f[2]; };

__device__ __forceinline__ unsigned short f2bf(float x) {
    unsigned u = __float_as_uint(x);
    unsigned r = (u + 0x7FFFu + ((u >> 16) & 1u)) >> 16;   // RNE
    return (unsigned short)r;
}
__device__ __forceinline__ float bf2f(unsigned short b) {
    return __uint_as_float(((unsigned)b) << 16);
}
__device__ __forceinline__ float sigf(float v)  { return 1.0f / (1.0f + __expf(-v)); }
__device__ __forceinline__ float tanhfast(float v) {
    // tanh(x) = 1 - 2/(exp(2x)+1); saturates correctly at +/-inf exp.
    return fmaf(-2.0f, 1.0f / (__expf(2.0f * v) + 1.0f), 1.0f);
}

// d_ws layout (dwords):
//   [0      , 8192 )  W_hh0 as B-matrix image (16 ntiles x 2 ksteps x 32 lanes x 8 dwords, bf16x2)
//   [8192   , 16384)  W_ih1 same
//   [16384  , 24576)  W_hh1 same
//   [24576  , 24832)  bias0 = b_ih0+b_hh0 padded to 256 (f32)
//   [24832  , 25088)  bias1 padded (f32)
//   [25088  , 25344)  W_ih0 column padded to 256 (f32)

__global__ void lstm_prep(const float* __restrict__ Wih0, const float* __restrict__ Whh0,
                          const float* __restrict__ bih0, const float* __restrict__ bhh0,
                          const float* __restrict__ Wih1, const float* __restrict__ Whh1,
                          const float* __restrict__ bih1, const float* __restrict__ bhh1,
                          unsigned* __restrict__ ws)
{
    int id = blockIdx.x * 256 + threadIdx.x;
    if (id < 3 * 8192) {
        int mat  = id / 8192;
        int e    = id % 8192;
        int v    = e & 7;             // dword within lane
        int lane = (e >> 3) & 31;
        int tk   = e >> 8;            // nt*2 + ks
        int ks   = tk & 1, nt = tk >> 1;
        int n    = nt * 16 + (lane & 15);           // padded gate column
        int gate = n >> 6, j = n & 63;              // gate block of 64 (i,f,g,o)
        int klo  = ks * 32 + (lane >> 4) * 8 + 2 * (v & 3) + (v >> 2) * 16;  // ISA 16-bit B layout (table order)
        const float* M = (mat == 0) ? Whh0 : (mat == 1) ? Wih1 : Whh1;  // [200][50]
        float v0 = 0.f, v1 = 0.f;
        if (j < 50) {
            int row = gate * 50 + j;
            if (klo     < 50) v0 = M[row * 50 + klo];
            if (klo + 1 < 50) v1 = M[row * 50 + klo + 1];
        }
        ws[id] = (unsigned)f2bf(v0) | ((unsigned)f2bf(v1) << 16);
    } else if (id < 3 * 8192 + 768) {
        int k = id - 3 * 8192;
        float* f = reinterpret_cast<float*>(ws);
        if (k < 256) {
            int gate = k >> 6, j = k & 63;
            f[24576 + k] = (j < 50) ? (bih0[gate * 50 + j] + bhh0[gate * 50 + j]) : 0.f;
        } else if (k < 512) {
            int kk = k - 256; int gate = kk >> 6, j = kk & 63;
            f[24832 + kk] = (j < 50) ? (bih1[gate * 50 + j] + bhh1[gate * 50 + j]) : 0.f;
        } else {
            int kk = k - 512; int gate = kk >> 6, j = kk & 63;
            f[25088 + kk] = (j < 50) ? Wih0[gate * 50 + j] : 0.f;  // W_ih0 is [200][1]
        }
    }
}

__global__ __launch_bounds__(128)
void lstm_main(const float* __restrict__ x, const unsigned* __restrict__ ws,
               const float* __restrict__ Wfc, const float* __restrict__ bfc,
               float* __restrict__ out)
{
    __shared__ float          xs[16 * 512];     // staged x tile, 32KB
    __shared__ float          preact[16 * 256]; // gate pre-activations, 16KB
    __shared__ unsigned short h0s[16 * 64];     // layer0 h (bf16, row-major, 64-col rows)
    __shared__ unsigned short h1s[16 * 64];     // layer1 h

    const int tid  = threadIdx.x;
    const int lane = tid & 31;
    const int warp = tid >> 5;          // 4 waves, wave w owns ntiles 4w..4w+3
    const int b0   = blockIdx.x * 16;

    for (int idx = tid; idx < 16 * 512; idx += 128) {
        int m = idx >> 9, t = idx & 511;
        xs[idx] = x[(b0 + m) * 512 + t];
    }
    for (int idx = tid; idx < 16 * 64; idx += 128) { h0s[idx] = 0; h1s[idx] = 0; }
    __syncthreads();

    // Hoist all B-matrices + bias columns into registers (loop-invariant).
    v16bf wHH0[4][2], wIH1[4][2], wHH1[4][2];
    float bias0v[4], bias1v[4], wih0v[4];
    const float* wf = reinterpret_cast<const float*>(ws);
#pragma unroll
    for (int q = 0; q < 4; ++q) {
        int nt = warp * 4 + q;
#pragma unroll
        for (int ks = 0; ks < 2; ++ks) {
            int tile = nt * 2 + ks;
            const v4f* p0 = reinterpret_cast<const v4f*>(ws)          + tile * 64 + lane * 2;
            const v4f* p1 = reinterpret_cast<const v4f*>(ws + 8192)   + tile * 64 + lane * 2;
            const v4f* p2 = reinterpret_cast<const v4f*>(ws + 16384)  + tile * 64 + lane * 2;
            BF16x16 u;
            u.f[0] = p0[0]; u.f[1] = p0[1]; wHH0[q][ks] = u.v;
            u.f[0] = p1[0]; u.f[1] = p1[1]; wIH1[q][ks] = u.v;
            u.f[0] = p2[0]; u.f[1] = p2[1]; wHH1[q][ks] = u.v;
        }
        int col = nt * 16 + (lane & 15);
        bias0v[q] = wf[24576 + col];
        bias1v[q] = wf[24832 + col];
        wih0v[q]  = wf[25088 + col];
    }

    float c0r[8], c1r[8];
#pragma unroll
    for (int e = 0; e < 8; ++e) { c0r[e] = 0.f; c1r[e] = 0.f; }
    const int cm  = tid >> 3;          // cell-update row (16 rows x 8 threads)
    const int cj0 = (tid & 7) * 8;     // cell-update col base (8 cells/thread)

    const int hi   = lane >> 4;        // A-matrix lane group (K sub-block select)
    const int am   = lane & 15;        // A-matrix row / C col-in-tile
    const int wrow = hi * 8;           // C-matrix row offset for this lane group

    for (int t = 0; t < 512; ++t) {
        // ---------- layer 0: gates = h0 @ Whh0^T  (+ x*Wih0 + bias) ----------
        BF16x16 a0, a1;
        {   // ISA 16-bit A layout: two 16B chunks at +0 and +32 bytes per lane
            const v4f* pa = reinterpret_cast<const v4f*>(&h0s[am * 64 + hi * 8]);
            a0.f[0] = pa[0]; a0.f[1] = pa[2];
            const v4f* pb = reinterpret_cast<const v4f*>(&h0s[am * 64 + 32 + hi * 8]);
            a1.f[0] = pb[0]; a1.f[1] = pb[2];
        }
        v8f acc[4];
#pragma unroll
        for (int q = 0; q < 4; ++q) {
            v8f c = {};
            c = __builtin_amdgcn_wmma_f32_16x16x32_bf16(false, a0.v, false, wHH0[q][0], (short)0, c, false, false);
            c = __builtin_amdgcn_wmma_f32_16x16x32_bf16(false, a1.v, false, wHH0[q][1], (short)0, c, false, false);
            acc[q] = c;
        }
#pragma unroll
        for (int q = 0; q < 4; ++q) {
            int col = (warp * 4 + q) * 16 + am;
#pragma unroll
            for (int r = 0; r < 8; ++r) {
                int M = r + wrow;
                preact[M * 256 + col] = acc[q][r] + bias0v[q] + xs[M * 512 + t] * wih0v[q];
            }
        }
        __syncthreads();

        // ---------- layer 0 cell update ----------
#pragma unroll
        for (int e = 0; e < 8; ++e) {
            int j = cj0 + e;
            float gi = preact[cm * 256 + j];
            float gf = preact[cm * 256 + 64 + j];
            float gg = preact[cm * 256 + 128 + j];
            float go = preact[cm * 256 + 192 + j];
            float c  = sigf(gf) * c0r[e] + sigf(gi) * tanhfast(gg);
            c0r[e] = c;
            h0s[cm * 64 + j] = f2bf(sigf(go) * tanhfast(c));
        }
        __syncthreads();

        // ---------- layer 1: gates = h1_new @ Wih1^T + h2_prev @ Whh1^T + bias ----------
        BF16x16 ai0, ai1, ar0, ar1;
        {
            const v4f* pa = reinterpret_cast<const v4f*>(&h0s[am * 64 + hi * 8]);
            ai0.f[0] = pa[0]; ai0.f[1] = pa[2];
            const v4f* pb = reinterpret_cast<const v4f*>(&h0s[am * 64 + 32 + hi * 8]);
            ai1.f[0] = pb[0]; ai1.f[1] = pb[2];
            const v4f* pc = reinterpret_cast<const v4f*>(&h1s[am * 64 + hi * 8]);
            ar0.f[0] = pc[0]; ar0.f[1] = pc[2];
            const v4f* pd = reinterpret_cast<const v4f*>(&h1s[am * 64 + 32 + hi * 8]);
            ar1.f[0] = pd[0]; ar1.f[1] = pd[2];
        }
#pragma unroll
        for (int q = 0; q < 4; ++q) {
            v8f c = {};
            c = __builtin_amdgcn_wmma_f32_16x16x32_bf16(false, ai0.v, false, wIH1[q][0], (short)0, c, false, false);
            c = __builtin_amdgcn_wmma_f32_16x16x32_bf16(false, ai1.v, false, wIH1[q][1], (short)0, c, false, false);
            c = __builtin_amdgcn_wmma_f32_16x16x32_bf16(false, ar0.v, false, wHH1[q][0], (short)0, c, false, false);
            c = __builtin_amdgcn_wmma_f32_16x16x32_bf16(false, ar1.v, false, wHH1[q][1], (short)0, c, false, false);
            acc[q] = c;
        }
#pragma unroll
        for (int q = 0; q < 4; ++q) {
            int col = (warp * 4 + q) * 16 + am;
#pragma unroll
            for (int r = 0; r < 8; ++r) {
                int M = r + wrow;
                preact[M * 256 + col] = acc[q][r] + bias1v[q];
            }
        }
        __syncthreads();

        // ---------- layer 1 cell update ----------
#pragma unroll
        for (int e = 0; e < 8; ++e) {
            int j = cj0 + e;
            float gi = preact[cm * 256 + j];
            float gf = preact[cm * 256 + 64 + j];
            float gg = preact[cm * 256 + 128 + j];
            float go = preact[cm * 256 + 192 + j];
            float c  = sigf(gf) * c1r[e] + sigf(gi) * tanhfast(gg);
            c1r[e] = c;
            h1s[cm * 64 + j] = f2bf(sigf(go) * tanhfast(c));
        }
        __syncthreads();
    }

    // ---------- final FC on h2[:, T-1, :] ----------
    if (tid < 16) {
        float s = 0.f;
        for (int j = 0; j < 50; ++j)
            s += bf2f(h1s[tid * 64 + j]) * Wfc[j];
        out[b0 + tid] = s + bfc[0];
    }
}

extern "C" void kernel_launch(void* const* d_in, const int* in_sizes, int n_in,
                              void* d_out, int out_size, void* d_ws, size_t ws_size,
                              hipStream_t stream)
{
    const float* x    = (const float*)d_in[0];
    const float* Wih0 = (const float*)d_in[1];
    const float* Whh0 = (const float*)d_in[2];
    const float* bih0 = (const float*)d_in[3];
    const float* bhh0 = (const float*)d_in[4];
    const float* Wih1 = (const float*)d_in[5];
    const float* Whh1 = (const float*)d_in[6];
    const float* bih1 = (const float*)d_in[7];
    const float* bhh1 = (const float*)d_in[8];
    const float* Wfc  = (const float*)d_in[9];
    const float* bfc  = (const float*)d_in[10];
    unsigned* ws = (unsigned*)d_ws;

    (void)in_sizes; (void)n_in; (void)out_size; (void)ws_size;

    lstm_prep<<<99, 256, 0, stream>>>(Wih0, Whh0, bih0, bhh0, Wih1, Whh1, bih1, bhh1, ws);
    lstm_main<<<256, 128, 0, stream>>>(x, ws, Wfc, bfc, (float*)d_out);
}
